// NATLayer3715_76261439308312
// MI455X (gfx1250) — compile-verified, matrix operand order checked by reading, not run on hardware
//
#include <hip/hip_runtime.h>
#include <hip/hip_bf16.h>
#include <math.h>

typedef __attribute__((ext_vector_type(16))) _Float16 v16h;
typedef __attribute__((ext_vector_type(8)))  _Float16 v8h;
typedef __attribute__((ext_vector_type(4)))  _Float16 v4h;
typedef __attribute__((ext_vector_type(8)))  float    v8f;
typedef __attribute__((ext_vector_type(4)))  int      v4i;

#define DIMC 128
#define HALFC 64
#define NHEADS 4
#define IMH 48
#define IMW 48
#define NB 2
#define NTOK (NB*IMH*IMW)   // 4608

// ---------------------------------------------------------------------------
// CDNA5 async-to-LDS path (guarded: falls back to VGPR round-trip if the
// builtin isn't declared by this toolchain).
// Probe-derived signature: (int4 addrspace(1)*, int4 addrspace(3)*, imm, imm)
// ---------------------------------------------------------------------------
#if defined(__gfx1250__) && __has_builtin(__builtin_amdgcn_global_load_async_to_lds_b128)
#define HAVE_ASYNC_LDS 1
typedef __attribute__((address_space(1))) v4i gas_v4i;
typedef __attribute__((address_space(3))) v4i las_v4i;
#else
#define HAVE_ASYNC_LDS 0
#endif

__device__ __forceinline__ void wait_async0() {
#if defined(__gfx1250__) && __has_builtin(__builtin_amdgcn_s_wait_asynccnt)
  __builtin_amdgcn_s_wait_asynccnt(0);
#elif defined(__gfx1250__)
  asm volatile("s_wait_asynccnt 0" ::: "memory");
#endif
}

// ---------------------------------------------------------------------------
// WMMA helpers (CDNA5 16x16x32 f16 -> f32)
// ---------------------------------------------------------------------------
__device__ __forceinline__ v8f wmma_fp16(v16h a, v16h b, v8f c) {
  return __builtin_amdgcn_wmma_f32_16x16x32_f16(false, a, false, b, (short)0, c,
                                                false, false);
}

union FragU { v16h v; struct { v8h lo, hi; } p; };

// Load a 16x32 f16 fragment (A layout; B uses same layout on W^T rows).
// Lane L (<16): row M=L, K chunks [0..7] and [16..23]; lane L+16: [8..15],[24..31].
__device__ __forceinline__ v16h load_frag(const _Float16* base, int row0, int ld, int k0) {
  int lane = threadIdx.x & 31;
  int r  = row0 + (lane & 15);
  int hi = lane >> 4;
  const _Float16* p = base + (size_t)r * ld + k0 + hi * 8;
  FragU u;
  u.p.lo = *(const v8h*)(p);
  u.p.hi = *(const v8h*)(p + 16);
  return u.v;
}

// ---------------------------------------------------------------------------
// Weight pack: [K][N] f32 row-major -> [N][K] f16 (B operand = rows of W^T)
// ---------------------------------------------------------------------------
__global__ void packT_kernel(const float* __restrict__ in, _Float16* __restrict__ out,
                             int K, int N) {
  int idx = blockIdx.x * blockDim.x + threadIdx.x;
  if (idx >= K * N) return;
  int n = idx / K, k = idx % K;
  out[idx] = (_Float16)in[k * N + n];
}

// ---------------------------------------------------------------------------
// LayerNorm over 128 channels; one wave per token. Optional residual add-in
// (x2 = x + y) written to outResid (f32); normalized result written f16.
// ---------------------------------------------------------------------------
__global__ void ln_kernel(const float* __restrict__ x, const float* __restrict__ addv,
                          const float* __restrict__ w, const float* __restrict__ b,
                          _Float16* __restrict__ outH, float* __restrict__ outResid,
                          int hasAdd) {
  int wave = threadIdx.x >> 5, lane = threadIdx.x & 31;
  int tok = blockIdx.x * (blockDim.x >> 5) + wave;
  if (tok >= NTOK) return;
  float4 v = *(const float4*)(x + (size_t)tok * DIMC + lane * 4);
  if (hasAdd) {
    float4 a = *(const float4*)(addv + (size_t)tok * DIMC + lane * 4);
    v.x += a.x; v.y += a.y; v.z += a.z; v.w += a.w;
  }
  if (outResid) *(float4*)(outResid + (size_t)tok * DIMC + lane * 4) = v;
  float s  = v.x + v.y + v.z + v.w;
  float s2 = v.x*v.x + v.y*v.y + v.z*v.z + v.w*v.w;
#pragma unroll
  for (int off = 16; off > 0; off >>= 1) {
    s  += __shfl_xor(s,  off, 32);
    s2 += __shfl_xor(s2, off, 32);
  }
  float m   = s * (1.0f / DIMC);
  float var = s2 * (1.0f / DIMC) - m * m;
  float inv = rsqrtf(var + 1e-5f);
  float4 wv = *(const float4*)(w + lane * 4);
  float4 bv = *(const float4*)(b + lane * 4);
  v4h o;
  o.x = (_Float16)((v.x - m) * inv * wv.x + bv.x);
  o.y = (_Float16)((v.y - m) * inv * wv.y + bv.y);
  o.z = (_Float16)((v.z - m) * inv * wv.z + bv.z);
  o.w = (_Float16)((v.w - m) * inv * wv.w + bv.w);
  *(v4h*)(outH + (size_t)tok * DIMC + lane * 4) = o;
}

// ---------------------------------------------------------------------------
// Implicit-GEMM 3x3 SAME conv + scale-shift-relu epilogue.
// Block = one row-of-16-tokens tile, CO/16 waves, one 16-wide N tile per wave.
// Patch (3 rows x 18 cols x CI halves) staged in LDS via async-to-LDS when
// available; A frags via ds_load_b128.
// OUT16: 1 -> f16 store, 0 -> f32 store/accumulate into outF.
// ---------------------------------------------------------------------------
template<int CI, int CO, int OUT16>
__global__ void conv3x3_kernel(const _Float16* __restrict__ X,
                               const _Float16* __restrict__ Wt,
                               const float* __restrict__ bias,
                               const float* __restrict__ gain,
                               const float* __restrict__ beta,
                               _Float16* __restrict__ outH,
                               float* __restrict__ outF, int accum) {
  constexpr int K9 = 9 * CI;
  int bid = blockIdx.x;
  int wt  = bid % (IMW / 16);
  int t   = bid / (IMW / 16);
  int h   = t % IMH;
  int bi  = t / IMH;
  int wbase = wt * 16;

  __shared__ __align__(16) _Float16 patch[3 * 18 * CI];
  int tid = threadIdx.x;

#if HAVE_ASYNC_LDS
  // 16B granulated cooperative halo stage, DMA-style (no VGPR round trip)
  {
    const int CHUNKS = 54 * CI / 8;
    for (int idx = tid; idx < CHUNKS; idx += blockDim.x) {
      int slab = idx / (CI / 8);
      int cc   = (idx % (CI / 8)) * 8;
      int r = slab / 18, col = slab % 18;
      int gh = h + r - 1, gw = wbase + col - 1;
      _Float16* dst = &patch[slab * CI + cc];
      if (gh >= 0 && gh < IMH && gw >= 0 && gw < IMW) {
        const _Float16* src = X + ((size_t)((bi * IMH + gh) * IMW + gw)) * CI + cc;
        __builtin_amdgcn_global_load_async_to_lds_b128(
            (gas_v4i*)src, (las_v4i*)dst, 0, 0);
      } else {
        *(v8h*)dst = (v8h)0;           // zero-fill SAME padding
      }
    }
    wait_async0();
  }
#else
  {
    const int CHUNKS = 54 * CI / 4;
    for (int idx = tid; idx < CHUNKS; idx += blockDim.x) {
      int slab = idx / (CI / 4);
      int cc   = (idx % (CI / 4)) * 4;
      int r = slab / 18, col = slab % 18;
      int gh = h + r - 1, gw = wbase + col - 1;
      unsigned long long val = 0ULL;
      if (gh >= 0 && gh < IMH && gw >= 0 && gw < IMW)
        val = *(const unsigned long long*)(X + ((size_t)((bi * IMH + gh) * IMW + gw)) * CI + cc);
      *(unsigned long long*)(&patch[slab * CI + cc]) = val;
    }
  }
#endif
  __syncthreads();

  int wave = tid >> 5, lane = tid & 31;
  int n0 = wave * 16;
  int wl = lane & 15, hi = lane >> 4;
  v8f acc = {};
#pragma unroll 4
  for (int k0 = 0; k0 < K9; k0 += 32) {
    int tap = k0 / CI, c0 = k0 % CI;
    int ky = tap / 3, kx = tap % 3;
    const _Float16* pa = &patch[(ky * 18 + (wl + kx)) * CI + c0 + hi * 8];
    FragU ua;
    ua.p.lo = *(const v8h*)(pa);
    ua.p.hi = *(const v8h*)(pa + 16);
    __builtin_prefetch(Wt + (size_t)(n0 + wl) * K9 + k0 + 64, 0, 3);
    v16h bfr = load_frag(Wt, n0, K9, k0);
    acc = wmma_fp16(ua.v, bfr, acc);
  }

  int ch = n0 + wl;
  float bb = bias[ch], gg = gain[ch], be = beta[ch];
#pragma unroll
  for (int i = 0; i < 8; i++) {
    int m = i + 8 * hi;                               // token within tile
    size_t tok = (size_t)((bi * IMH + h) * IMW + wbase + m);
    float v = (acc[i] + bb) * gg + be;
    v = fmaxf(v, 0.0f);                               // relu
    if (OUT16) {
      outH[tok * CO + ch] = (_Float16)v;
    } else {
      if (accum) outF[tok * CO + ch] += v;
      else       outF[tok * CO + ch]  = v;
    }
  }
}

// ---------------------------------------------------------------------------
// Generic GEMM: C[M][N] = epi(A[M][K]f16 @ Wt[N][K]f16 + bias)
// Each wave owns a 32M x 16N tile: two WMMAs share one B fragment
// (1.5 fragment loads per WMMA instead of 2).
// EPI 0: f32 out (qkv)   1: f16 out (proj)
// EPI 2: GELU(exact) f16 out (fc1)   3: +residual, f32 out (fc2 -> d_out)
// Block 128 thr = 4 waves covering a 32M x 64N tile.
// ---------------------------------------------------------------------------
template<int EPI>
__global__ void gemm_kernel(const _Float16* __restrict__ A,
                            const _Float16* __restrict__ Wt,
                            const float* __restrict__ bias,
                            const float* __restrict__ res,
                            float* __restrict__ outF,
                            _Float16* __restrict__ outH,
                            int M, int N, int K) {
  int wave = threadIdx.x >> 5, lane = threadIdx.x & 31;
  int m0 = blockIdx.x * 32;
  int n0 = blockIdx.y * 64 + wave * 16;
  if (m0 >= M || n0 >= N) return;
  v8f acc0 = {}, acc1 = {};
#pragma unroll 4
  for (int k0 = 0; k0 < K; k0 += 32) {
    __builtin_prefetch(Wt + (size_t)(n0 + (lane & 15)) * K + k0 + 64, 0, 3);
    v16h b  = load_frag(Wt, n0, K, k0);
    v16h a0 = load_frag(A,  m0,      K, k0);
    v16h a1 = load_frag(A,  m0 + 16, K, k0);
    acc0 = wmma_fp16(a0, b, acc0);
    acc1 = wmma_fp16(a1, b, acc1);
  }
  int l = lane & 15, hi = lane >> 4;
  int n = n0 + l;
  float bv = bias[n];
#pragma unroll
  for (int half = 0; half < 2; half++) {
    v8f acc = half ? acc1 : acc0;
    int mb = m0 + half * 16;
#pragma unroll
    for (int i = 0; i < 8; i++) {
      int m = mb + i + 8 * hi;
      float v = acc[i] + bv;
      if (EPI == 2) v = 0.5f * v * (1.0f + erff(v * 0.70710678118654752f));
      if (EPI == 3) v += res[(size_t)m * N + n];
      if (EPI == 0 || EPI == 3) outF[(size_t)m * N + n] = v;
      else                      outH[(size_t)m * N + n] = (_Float16)v;
    }
  }
}

// ---------------------------------------------------------------------------
// Neighborhood attention: one wave per (token, head). d=16, scale=0.25.
// qkv layout [tok][192] = (s, head, d). Softmax in-wave via shfl_xor.
// ---------------------------------------------------------------------------
template<int KSZ>
__global__ void nat_kernel(const float* __restrict__ qkv,
                           const float* __restrict__ rpb,
                           _Float16* __restrict__ out) {
  constexpr int NW = KSZ * KSZ;
  constexpr int PL = (NW + 31) / 32;
  constexpr int R  = 2 * KSZ - 1;
  int wave = threadIdx.x >> 5, lane = threadIdx.x & 31;
  int gid  = blockIdx.x * (blockDim.x >> 5) + wave;
  int head = gid & 3;
  int tok  = gid >> 2;
  if (tok >= NTOK) return;
  int j = tok % IMW; int t = tok / IMW; int i = t % IMH; int bi = t / IMH;

  const float* qp = qkv + (size_t)tok * 192 + head * 16;
  float4 q0 = *(const float4*)(qp);
  float4 q1 = *(const float4*)(qp + 4);
  float4 q2 = *(const float4*)(qp + 8);
  float4 q3 = *(const float4*)(qp + 12);
  const float sc = 0.25f;   // 16^-0.5
  q0.x*=sc;q0.y*=sc;q0.z*=sc;q0.w*=sc; q1.x*=sc;q1.y*=sc;q1.z*=sc;q1.w*=sc;
  q2.x*=sc;q2.y*=sc;q2.z*=sc;q2.w*=sc; q3.x*=sc;q3.y*=sc;q3.z*=sc;q3.w*=sc;

  int ih0 = min(max(i - KSZ / 2, 0), IMH - KSZ);
  int iw0 = min(max(j - KSZ / 2, 0), IMW - KSZ);

  float logit[PL];
  float mx = -1e30f;
#pragma unroll
  for (int p = 0; p < PL; p++) {
    int n = lane + 32 * p;
    float lg = -1e30f;
    if (n < NW) {
      int wy = n / KSZ, wx = n % KSZ;
      int gh = ih0 + wy, gw = iw0 + wx;
      const float* kp = qkv + (size_t)((bi * IMH + gh) * IMW + gw) * 192 + 64 + head * 16;
      float4 k0 = *(const float4*)(kp);
      float4 k1 = *(const float4*)(kp + 4);
      float4 k2 = *(const float4*)(kp + 8);
      float4 k3 = *(const float4*)(kp + 12);
      float s = q0.x*k0.x + q0.y*k0.y + q0.z*k0.z + q0.w*k0.w
              + q1.x*k1.x + q1.y*k1.y + q1.z*k1.z + q1.w*k1.w
              + q2.x*k2.x + q2.y*k2.y + q2.z*k2.z + q2.w*k2.w
              + q3.x*k3.x + q3.y*k3.y + q3.z*k3.z + q3.w*k3.w;
      s += rpb[(size_t)(head * R + (gh - i + KSZ - 1)) * R + (gw - j + KSZ - 1)];
      lg = s;
    }
    logit[p] = lg;
    mx = fmaxf(mx, lg);
  }
#pragma unroll
  for (int off = 16; off > 0; off >>= 1) mx = fmaxf(mx, __shfl_xor(mx, off, 32));

  float se = 0.0f;
#pragma unroll
  for (int p = 0; p < PL; p++) {
    float e = (logit[p] > -1e29f) ? __expf(logit[p] - mx) : 0.0f;
    logit[p] = e;
    se += e;
  }
#pragma unroll
  for (int off = 16; off > 0; off >>= 1) se += __shfl_xor(se, off, 32);
  float inv = 1.0f / se;

  float4 o0 = {0,0,0,0}, o1 = {0,0,0,0}, o2 = {0,0,0,0}, o3 = {0,0,0,0};
#pragma unroll
  for (int p = 0; p < PL; p++) {
    int n = lane + 32 * p;
    if (n < NW) {
      int wy = n / KSZ, wx = n % KSZ;
      int gh = ih0 + wy, gw = iw0 + wx;
      const float* vp = qkv + (size_t)((bi * IMH + gh) * IMW + gw) * 192 + 128 + head * 16;
      float e = logit[p];
      float4 v0 = *(const float4*)(vp);
      float4 v1 = *(const float4*)(vp + 4);
      float4 v2 = *(const float4*)(vp + 8);
      float4 v3 = *(const float4*)(vp + 12);
      o0.x += e*v0.x; o0.y += e*v0.y; o0.z += e*v0.z; o0.w += e*v0.w;
      o1.x += e*v1.x; o1.y += e*v1.y; o1.z += e*v1.z; o1.w += e*v1.w;
      o2.x += e*v2.x; o2.y += e*v2.y; o2.z += e*v2.z; o2.w += e*v2.w;
      o3.x += e*v3.x; o3.y += e*v3.y; o3.z += e*v3.z; o3.w += e*v3.w;
    }
  }
  float ov[16] = {o0.x,o0.y,o0.z,o0.w, o1.x,o1.y,o1.z,o1.w,
                  o2.x,o2.y,o2.z,o2.w, o3.x,o3.y,o3.z,o3.w};
#pragma unroll
  for (int d = 0; d < 16; d++) {
#pragma unroll
    for (int off = 16; off > 0; off >>= 1) ov[d] += __shfl_xor(ov[d], off, 32);
  }
  if (lane < 16) {
    float sel = 0.0f;
#pragma unroll
    for (int d = 0; d < 16; d++) if (lane == d) sel = ov[d];
    out[(size_t)tok * 64 + head * 16 + lane] = (_Float16)(sel * inv);
  }
}

// ---------------------------------------------------------------------------
// Host orchestration
// ---------------------------------------------------------------------------
static inline size_t alignup(size_t x) { return (x + 255) & ~(size_t)255; }

extern "C" void kernel_launch(void* const* d_in, const int* in_sizes, int n_in,
                              void* d_out, int out_size, void* d_ws, size_t ws_size,
                              hipStream_t stream) {
  (void)in_sizes; (void)n_in; (void)out_size; (void)ws_size;
  const float* x   = (const float*)d_in[0];
  const float* n1w = (const float*)d_in[1];
  const float* n1b = (const float*)d_in[2];
  const float *cw[6], *cb[6], *cg[6], *cbe[6];
  for (int i = 0; i < 6; i++) {        // conv1, conv2, conv22, conv3, conv4, conv33
    cw[i]  = (const float*)d_in[3 + 4 * i];
    cb[i]  = (const float*)d_in[4 + 4 * i];
    cg[i]  = (const float*)d_in[5 + 4 * i];
    cbe[i] = (const float*)d_in[6 + 4 * i];
  }
  const float *aqw[3], *aqb[3], *apw[3], *apb[3], *arp[3];  // attn7, attn3, attn15
  for (int i = 0; i < 3; i++) {
    aqw[i] = (const float*)d_in[27 + 5 * i];
    aqb[i] = (const float*)d_in[28 + 5 * i];
    apw[i] = (const float*)d_in[29 + 5 * i];
    apb[i] = (const float*)d_in[30 + 5 * i];
    arp[i] = (const float*)d_in[31 + 5 * i];
  }
  const float* n2w  = (const float*)d_in[42];
  const float* n2b  = (const float*)d_in[43];
  const float* fc1w = (const float*)d_in[44];
  const float* fc1b = (const float*)d_in[45];
  const float* fc2w = (const float*)d_in[46];
  const float* fc2b = (const float*)d_in[47];

  // -------- workspace carve --------
  char* w = (char*)d_ws;
  auto take = [&](size_t bytes) { char* p = w; w += alignup(bytes); return p; };
  _Float16* xn16   = (_Float16*)take((size_t)NTOK * 128 * 2);
  _Float16* wtc1   = (_Float16*)take((size_t)64 * 1152 * 2);
  _Float16* wtc2   = (_Float16*)take((size_t)64 * 1152 * 2);
  _Float16* wtc22  = (_Float16*)take((size_t)64 * 1152 * 2);
  _Float16* wtc3   = (_Float16*)take((size_t)128 * 576 * 2);
  _Float16* wtc4   = (_Float16*)take((size_t)128 * 576 * 2);
  _Float16* wtc33  = (_Float16*)take((size_t)128 * 576 * 2);
  _Float16* wtq[3]; _Float16* wtp[3];
  for (int i = 0; i < 3; i++) {
    wtq[i] = (_Float16*)take((size_t)192 * 64 * 2);
    wtp[i] = (_Float16*)take((size_t)64 * 64 * 2);
  }
  _Float16* wtf1   = (_Float16*)take((size_t)512 * 128 * 2);
  _Float16* wtf2   = (_Float16*)take((size_t)128 * 512 * 2);
  _Float16* x15h   = (_Float16*)take((size_t)NTOK * 64 * 2);
  _Float16* x7h    = (_Float16*)take((size_t)NTOK * 64 * 2);
  _Float16* x3h    = (_Float16*)take((size_t)NTOK * 64 * 2);
  float*    qkvF   = (float*)   take((size_t)NTOK * 192 * 4);
  _Float16* attno  = (_Float16*)take((size_t)NTOK * 64 * 2);
  _Float16* projo  = (_Float16*)take((size_t)NTOK * 64 * 2);
  float*    yF     = (float*)   take((size_t)NTOK * 128 * 4);
  float*    x2F    = (float*)   take((size_t)NTOK * 128 * 4);
  _Float16* xn2h   = (_Float16*)take((size_t)NTOK * 128 * 2);
  _Float16* h1h    = (_Float16*)take((size_t)NTOK * 512 * 2);

  auto pack = [&](const float* in, _Float16* out, int K, int N) {
    int total = K * N;
    packT_kernel<<<(total + 255) / 256, 256, 0, stream>>>(in, out, K, N);
  };
  pack(cw[0], wtc1, 1152, 64);  pack(cw[1], wtc2, 1152, 64);  pack(cw[2], wtc22, 1152, 64);
  pack(cw[3], wtc3,  576, 128); pack(cw[4], wtc4,  576, 128); pack(cw[5], wtc33, 576, 128);
  for (int i = 0; i < 3; i++) { pack(aqw[i], wtq[i], 64, 192); pack(apw[i], wtp[i], 64, 64); }
  pack(fc1w, wtf1, 128, 512);   pack(fc2w, wtf2, 512, 128);

  const int LNB = NTOK / 8;          // 576 blocks, 8 waves each
  ln_kernel<<<LNB, 256, 0, stream>>>(x, nullptr, n1w, n1b, xn16, nullptr, 0);

  const int CVB = NB * IMH * (IMW / 16);   // 288 tiles
  // stage-1 convs (128 -> 64), f16 out
  conv3x3_kernel<128, 64, 1><<<CVB, 128, 0, stream>>>(xn16, wtc1,  cb[0], cg[0], cbe[0], x15h, nullptr, 0);
  conv3x3_kernel<128, 64, 1><<<CVB, 128, 0, stream>>>(xn16, wtc2,  cb[1], cg[1], cbe[1], x7h,  nullptr, 0);
  conv3x3_kernel<128, 64, 1><<<CVB, 128, 0, stream>>>(xn16, wtc22, cb[2], cg[2], cbe[2], x3h,  nullptr, 0);

  const dim3 gQKV(NTOK / 32, 3), gP(NTOK / 32, 1);
  const int NATB = (NTOK * NHEADS) / 8;    // 2304 blocks

  // branch 15: conv1 -> attn15(idx 2) -> conv3 (cw[3]), y := result
  gemm_kernel<0><<<gQKV, 128, 0, stream>>>(x15h, wtq[2], aqb[2], nullptr, qkvF, nullptr, NTOK, 192, 64);
  nat_kernel<15><<<NATB, 256, 0, stream>>>(qkvF, arp[2], attno);
  gemm_kernel<1><<<gP, 128, 0, stream>>>(attno, wtp[2], apb[2], nullptr, nullptr, projo, NTOK, 64, 64);
  conv3x3_kernel<64, 128, 0><<<CVB, 256, 0, stream>>>(projo, wtc3, cb[3], cg[3], cbe[3], nullptr, yF, 0);

  // branch 7: conv2 -> attn7(idx 0) -> conv4 (cw[4]), y += result
  gemm_kernel<0><<<gQKV, 128, 0, stream>>>(x7h, wtq[0], aqb[0], nullptr, qkvF, nullptr, NTOK, 192, 64);
  nat_kernel<7><<<NATB, 256, 0, stream>>>(qkvF, arp[0], attno);
  gemm_kernel<1><<<gP, 128, 0, stream>>>(attno, wtp[0], apb[0], nullptr, nullptr, projo, NTOK, 64, 64);
  conv3x3_kernel<64, 128, 0><<<CVB, 256, 0, stream>>>(projo, wtc4, cb[4], cg[4], cbe[4], nullptr, yF, 1);

  // branch 3: conv22 -> attn3(idx 1) -> conv33 (cw[5]), y += result
  gemm_kernel<0><<<gQKV, 128, 0, stream>>>(x3h, wtq[1], aqb[1], nullptr, qkvF, nullptr, NTOK, 192, 64);
  nat_kernel<3><<<NATB, 256, 0, stream>>>(qkvF, arp[1], attno);
  gemm_kernel<1><<<gP, 128, 0, stream>>>(attno, wtp[1], apb[1], nullptr, nullptr, projo, NTOK, 64, 64);
  conv3x3_kernel<64, 128, 0><<<CVB, 256, 0, stream>>>(projo, wtc33, cb[5], cg[5], cbe[5], nullptr, yF, 1);

  // x2 = x + y; LN2 -> f16
  ln_kernel<<<LNB, 256, 0, stream>>>(x, yF, n2w, n2b, xn2h, x2F, 1);

  // fc1 (GELU exact) -> f16 ; fc2 + residual -> d_out (f32)
  gemm_kernel<2><<<dim3(NTOK / 32, 8), 128, 0, stream>>>(xn2h, wtf1, fc1b, nullptr, nullptr, h1h, NTOK, 512, 128);
  gemm_kernel<3><<<dim3(NTOK / 32, 2), 128, 0, stream>>>(h1h, wtf2, fc2b, x2F, (float*)d_out, nullptr, NTOK, 128, 512);
}
